// KAN_32882269618299
// MI455X (gfx1250) — matrix operand order repaired and checked by reading, not compiled, not run on hardware
//
#include <hip/hip_runtime.h>
#include <hip/hip_bf16.h>

typedef __attribute__((ext_vector_type(16))) _Float16 v16h;
typedef __attribute__((ext_vector_type(8)))  _Float16 v8h;
typedef __attribute__((ext_vector_type(8)))  float    v8f;

#define D128 128

__device__ __forceinline__ void lds_sync_wave() {
    // DS ops are in-order within a wave; waiting DScnt==0 makes prior LDS
    // stores visible to all lanes of this wave. "memory" stops reordering.
    asm volatile("s_wait_dscnt 0" ::: "memory");
}

__device__ __forceinline__ v8f zero8() {
    v8f z = {0.f, 0.f, 0.f, 0.f, 0.f, 0.f, 0.f, 0.f};
    return z;
}

// ---- fragment loaders (wave32 WMMA 16x16x32 f16 layouts, ISA 7.12.2) ----

// A: 16xK f16 buffer (LDS), row-major, row stride ldh halfs.
// lane L: M = L&15, half h = L>>4; elems 0..7 = K[kb+8h .. +7], 8..15 = K[kb+16+8h .. +7]
__device__ __forceinline__ v16h a_frag_lds(const _Float16* buf, int ldh, int kb, int lane) {
    const int m = lane & 15, h = (lane >> 4) & 1;
    const _Float16* p0 = buf + m * ldh + kb + h * 8;
    union { v16h v; v8h u[2]; } r;
    r.u[0] = *(const v8h*)p0;
    r.u[1] = *(const v8h*)(p0 + 16);
    return r.v;
}

// A from a per-lane global fp32 row (convert to f16 on the fly)
__device__ __forceinline__ v16h a_frag_f32row(const float* row, int kb, int lane) {
    const int h = (lane >> 4) & 1;
    const float* p0 = row + kb + h * 8;
    const float* p1 = p0 + 16;
    v16h a;
#pragma unroll
    for (int i = 0; i < 8; ++i) a[i] = (_Float16)p0[i];
#pragma unroll
    for (int i = 0; i < 8; ++i) a[8 + i] = (_Float16)p1[i];
    return a;
}

// B: weights stored transposed f16 [N][Ktot]; lane L: N = nb+(L&15),
// elems 0..7 = K[kb+8h .. +7], 8..15 = K[kb+16+8h .. +7]
__device__ __forceinline__ v16h b_frag(const _Float16* wt, int Ktot, int nb, int kb, int lane) {
    const int n = lane & 15, h = (lane >> 4) & 1;
    const _Float16* p = wt + (nb + n) * Ktot + kb + h * 8;
    union { v16h v; v8h u[2]; } r;
    r.u[0] = *(const v8h*)p;
    r.u[1] = *(const v8h*)(p + 16);
    return r.v;
}

// one K-step against all 8 N-tiles. Preload ALL B fragments first so the
// 16 b128 loads batch into clauses and overlap the 8 WMMAs (instead of a
// load -> s_wait_loadcnt 0 -> wmma lockstep).
__device__ __forceinline__ void wmma_row(v8f acc[8], v16h a, const _Float16* wt,
                                         int Ktot, int kb, int lane) {
    v16h b[8];
#pragma unroll
    for (int n = 0; n < 8; ++n) b[n] = b_frag(wt, Ktot, n * 16, kb, lane);
#pragma unroll
    for (int n = 0; n < 8; ++n)
        acc[n] = __builtin_amdgcn_wmma_f32_16x16x32_f16(
            false, a, false, b[n], (short)0, acc[n], false, false);
}

// GroupNorm(groups=1) over the 128 channels of each of the 16 rows held in
// acc[8] (C-layout: lane L holds col n*16+(L&15), rows v+8*(L>>4)).
__device__ __forceinline__ void groupnorm_acc(v8f acc[8], const float* gamma,
                                              const float* beta, int lane, bool relu) {
    const int c0 = lane & 15;
    float g[8], bt[8];
#pragma unroll
    for (int n = 0; n < 8; ++n) { g[n] = gamma[n * 16 + c0]; bt[n] = beta[n * 16 + c0]; }
#pragma unroll
    for (int v = 0; v < 8; ++v) {
        float s = 0.f, q = 0.f;
#pragma unroll
        for (int n = 0; n < 8; ++n) { float x = acc[n][v]; s += x; q += x * x; }
#pragma unroll
        for (int msk = 1; msk <= 8; msk <<= 1) {
            s += __shfl_xor(s, msk, 32);
            q += __shfl_xor(q, msk, 32);
        }
        float mean = s * 0.0078125f;
        float var  = q * 0.0078125f - mean * mean;
        float rs = rsqrtf(var + 1e-5f);
#pragma unroll
        for (int n = 0; n < 8; ++n) {
            float x = (acc[n][v] - mean) * rs * g[n] + bt[n];
            acc[n][v] = relu ? fmaxf(x, 0.f) : x;
        }
    }
}

__device__ __forceinline__ void store_acc_f16(const v8f acc[8], _Float16* buf,
                                              int ldh, int coff, int lane) {
    const int c0 = lane & 15, h = (lane >> 4) & 1;
#pragma unroll
    for (int v = 0; v < 8; ++v) {
        const int row = v + 8 * h;
#pragma unroll
        for (int n = 0; n < 8; ++n)
            buf[row * ldh + coff + n * 16 + c0] = (_Float16)acc[n][v];
    }
}

// ---------------- weight prep: fp32 [K][N] -> f16 [N][K] ----------------
__global__ void kan_wprep(const float* __restrict__ src, _Float16* __restrict__ dst,
                          int K, int N) {
    int i = blockIdx.x * blockDim.x + threadIdx.x;
    if (i >= K * N) return;
    int k = i / N, n = i - k * N;
    dst[n * K + k] = (_Float16)src[k * N + n];
}

// ---------------- a0 = agts @ W_a ----------------
__global__ __launch_bounds__(128) void kan_node_pre(
    const float* __restrict__ agts, const _Float16* __restrict__ Wt_a,
    float* __restrict__ a0, int N) {
    const int lane = threadIdx.x & 31, wv = threadIdx.x >> 5;
    const int r0 = (blockIdx.x * 4 + wv) * 16;
    if (r0 >= N) return;
    const int m = lane & 15, h = (lane >> 4) & 1, c0 = m;
    int rowm = r0 + m; if (rowm >= N) rowm = N - 1;
    const float* arow = agts + rowm * D128;
    v8f acc[8];
#pragma unroll
    for (int n = 0; n < 8; ++n) acc[n] = zero8();
#pragma unroll
    for (int kb = 0; kb < 128; kb += 32)
        wmma_row(acc, a_frag_f32row(arow, kb, lane), Wt_a, 128, kb, lane);
#pragma unroll
    for (int v = 0; v < 8; ++v) {
        const int row = r0 + v + 8 * h;
        if (row < N) {
#pragma unroll
            for (int n = 0; n < 8; ++n) a0[row * D128 + n * 16 + c0] = acc[n][v];
        }
    }
}

// ---------------- edge pipeline + scatter ----------------
__global__ __launch_bounds__(128) void kan_edge(
    const float* __restrict__ agts, const float* __restrict__ ctx,
    const float* __restrict__ agt_ctrs, const float* __restrict__ ctx_ctrs,
    const int* __restrict__ hi, const int* __restrict__ wi,
    const float* __restrict__ W_d1, const float* __restrict__ b_d1,
    const _Float16* __restrict__ Wt_d2, const float* __restrict__ g_d2, const float* __restrict__ b_d2,
    const _Float16* __restrict__ Wt_q,  const float* __restrict__ g_q,  const float* __restrict__ b_q,
    const _Float16* __restrict__ Wt_c1, const float* __restrict__ g_c1, const float* __restrict__ b_c1,
    const _Float16* __restrict__ Wt_c2,
    float* __restrict__ a0, int E) {
    __shared__ _Float16 s_tmp[4][16 * 128];   // d1, then c1 mid result
    __shared__ _Float16 s_cat[4][16 * 256];   // [d | q] halves of concat
    __shared__ int s_hi[4][16];
    __shared__ int s_wi[4][16];

    const int lane = threadIdx.x & 31, wv = threadIdx.x >> 5;
    const int e0 = (blockIdx.x * 4 + wv) * 16;
    if (e0 >= E) return;
    _Float16* tmp = s_tmp[wv];
    _Float16* cat = s_cat[wv];
    int* his = s_hi[wv];
    int* wis = s_wi[wv];
    const int m = lane & 15, h = (lane >> 4) & 1, c0 = m;

    if (lane < 16) {
        int e = e0 + lane; if (e >= E) e = E - 1;
        his[lane] = hi[e];
        wis[lane] = wi[e];
    }
    lds_sync_wave();

    // ---- dist embed layer 1: relu(dist @ W_d1 + b) -> tmp (f16) ----
    {
        const int hm = his[m], wm = wis[m];
        const float dx = agt_ctrs[2 * hm]     - ctx_ctrs[2 * wm];
        const float dy = agt_ctrs[2 * hm + 1] - ctx_ctrs[2 * wm + 1];
#pragma unroll 8
        for (int c = 0; c < 64; ++c) {
            const int col = h * 64 + c;
            float v = fmaxf(dx * W_d1[col] + dy * W_d1[128 + col] + b_d1[col], 0.f);
            tmp[m * 128 + col] = (_Float16)v;
        }
    }
    lds_sync_wave();

    // ---- d = relu(GN(d1 @ W_d2)) -> cat[:,0:128] ----
    {
        v8f acc[8];
#pragma unroll
        for (int n = 0; n < 8; ++n) acc[n] = zero8();
#pragma unroll
        for (int kb = 0; kb < 128; kb += 32)
            wmma_row(acc, a_frag_lds(tmp, 128, kb, lane), Wt_d2, 128, kb, lane);
        groupnorm_acc(acc, g_d2, b_d2, lane, true);
        store_acc_f16(acc, cat, 256, 0, lane);
    }

    // ---- q = relu(GN(agts[hi] @ W_q)) -> cat[:,128:256] ----
    {
        const float* arow = agts + his[m] * D128;
        v8f acc[8];
#pragma unroll
        for (int n = 0; n < 8; ++n) acc[n] = zero8();
#pragma unroll
        for (int kb = 0; kb < 128; kb += 32)
            wmma_row(acc, a_frag_f32row(arow, kb, lane), Wt_q, 128, kb, lane);
        groupnorm_acc(acc, g_q, b_q, lane, true);
        store_acc_f16(acc, cat, 256, 128, lane);
    }
    lds_sync_wave();

    // ---- mid = relu(GN(concat(d,q,ctx[wi]) @ W_c1)) -> tmp ----
    {
        const float* crow = ctx + wis[m] * D128;
        v8f acc[8];
#pragma unroll
        for (int n = 0; n < 8; ++n) acc[n] = zero8();
#pragma unroll
        for (int kb = 0; kb < 128; kb += 32)   // ctx from global (L2): start first
            wmma_row(acc, a_frag_f32row(crow, kb, lane), Wt_c1, 384, kb + 256, lane);
#pragma unroll
        for (int kb = 0; kb < 256; kb += 32)   // d,q from LDS
            wmma_row(acc, a_frag_lds(cat, 256, kb, lane), Wt_c1, 384, kb, lane);
        groupnorm_acc(acc, g_c1, b_c1, lane, true);
        store_acc_f16(acc, tmp, 128, 0, lane);
    }
    lds_sync_wave();

    // ---- msg = mid @ W_c2 ; scatter-add into a0[hi] ----
    {
        v8f acc[8];
#pragma unroll
        for (int n = 0; n < 8; ++n) acc[n] = zero8();
#pragma unroll
        for (int kb = 0; kb < 128; kb += 32)
            wmma_row(acc, a_frag_lds(tmp, 128, kb, lane), Wt_c2, 128, kb, lane);
#pragma unroll
        for (int v = 0; v < 8; ++v) {
            const int row = v + 8 * h;
            const int e = e0 + row;
            if (e < E) {
                float* dst = a0 + his[row] * D128 + c0;
#pragma unroll
                for (int n = 0; n < 8; ++n) atomicAdd(dst + n * 16, acc[n][v]);
            }
        }
    }
}

// ---------------- post: GN+relu(a0) @ W_l -> GN -> +res -> relu ----------------
__global__ __launch_bounds__(128) void kan_node_post(
    const float* __restrict__ agts, const float* __restrict__ a0,
    const _Float16* __restrict__ Wt_l,
    const float* __restrict__ g_n, const float* __restrict__ b_n,
    const float* __restrict__ g_l, const float* __restrict__ b_l,
    float* __restrict__ out, int N) {
    __shared__ _Float16 s_x[4][16 * 128];
    const int lane = threadIdx.x & 31, wv = threadIdx.x >> 5;
    const int r0 = (blockIdx.x * 4 + wv) * 16;
    if (r0 >= N) return;
    _Float16* xb = s_x[wv];
    const int m = lane & 15, h = (lane >> 4) & 1, c0 = m;

    // per-row GN + relu of scattered accumulator; two lanes share a row
    {
        int row = r0 + m; if (row >= N) row = N - 1;
        const float* src = a0 + row * D128 + h * 64;
        float s = 0.f, q = 0.f;
#pragma unroll 8
        for (int c = 0; c < 64; ++c) { float x = src[c]; s += x; q += x * x; }
        s += __shfl_xor(s, 16, 32);
        q += __shfl_xor(q, 16, 32);
        const float mean = s * 0.0078125f;
        const float var  = q * 0.0078125f - mean * mean;
        const float rs = rsqrtf(var + 1e-5f);
#pragma unroll 8
        for (int c = 0; c < 64; ++c) {
            const int col = h * 64 + c;
            float x = fmaxf((src[c] - mean) * rs * g_n[col] + b_n[col], 0.f);
            xb[m * 128 + col] = (_Float16)x;
        }
    }
    lds_sync_wave();

    v8f acc[8];
#pragma unroll
    for (int n = 0; n < 8; ++n) acc[n] = zero8();
#pragma unroll
    for (int kb = 0; kb < 128; kb += 32)
        wmma_row(acc, a_frag_lds(xb, 128, kb, lane), Wt_l, 128, kb, lane);
    groupnorm_acc(acc, g_l, b_l, lane, false);

#pragma unroll
    for (int v = 0; v < 8; ++v) {
        const int row = r0 + v + 8 * h;
        if (row < N) {
            const float* rr = agts + row * D128;
#pragma unroll
            for (int n = 0; n < 8; ++n) {
                const int col = n * 16 + c0;
                out[row * D128 + col] = fmaxf(acc[n][v] + rr[col], 0.f);
            }
        }
    }
}

extern "C" void kernel_launch(void* const* d_in, const int* in_sizes, int n_in,
                              void* d_out, int out_size, void* d_ws, size_t ws_size,
                              hipStream_t stream) {
    (void)n_in; (void)out_size; (void)ws_size;
    const float* agts     = (const float*)d_in[0];
    const float* ctx      = (const float*)d_in[1];
    const float* agt_ctrs = (const float*)d_in[2];
    const float* ctx_ctrs = (const float*)d_in[3];
    const int*   hi       = (const int*)d_in[4];
    const int*   wi       = (const int*)d_in[5];
    const float* W_d1 = (const float*)d_in[6];
    const float* b_d1 = (const float*)d_in[7];
    const float* W_d2 = (const float*)d_in[8];
    const float* g_d2 = (const float*)d_in[9];
    const float* b_d2 = (const float*)d_in[10];
    const float* W_q  = (const float*)d_in[11];
    const float* g_q  = (const float*)d_in[12];
    const float* b_q  = (const float*)d_in[13];
    const float* W_c1 = (const float*)d_in[14];
    const float* g_c1 = (const float*)d_in[15];
    const float* b_c1 = (const float*)d_in[16];
    const float* W_c2 = (const float*)d_in[17];
    const float* W_a  = (const float*)d_in[18];
    const float* g_n  = (const float*)d_in[19];
    const float* b_n  = (const float*)d_in[20];
    const float* W_l  = (const float*)d_in[21];
    const float* g_l  = (const float*)d_in[22];
    const float* b_l  = (const float*)d_in[23];

    const int N = in_sizes[0] / D128;
    const int E = in_sizes[4];

    char* ws = (char*)d_ws;
    _Float16* Wt_d2 = (_Float16*)(ws + 0);
    _Float16* Wt_q  = (_Float16*)(ws + 32768);
    _Float16* Wt_c1 = (_Float16*)(ws + 65536);     // 384*128*2 = 98304
    _Float16* Wt_c2 = (_Float16*)(ws + 163840);
    _Float16* Wt_a  = (_Float16*)(ws + 196608);
    _Float16* Wt_l  = (_Float16*)(ws + 229376);
    float*    a0    = (float*)(ws + 262144);       // N*128*4

    // weight transpose/convert to f16 [N][K]
    kan_wprep<<<(128 * 128 + 255) / 256, 256, 0, stream>>>(W_d2, Wt_d2, 128, 128);
    kan_wprep<<<(128 * 128 + 255) / 256, 256, 0, stream>>>(W_q,  Wt_q,  128, 128);
    kan_wprep<<<(384 * 128 + 255) / 256, 256, 0, stream>>>(W_c1, Wt_c1, 384, 128);
    kan_wprep<<<(128 * 128 + 255) / 256, 256, 0, stream>>>(W_c2, Wt_c2, 128, 128);
    kan_wprep<<<(128 * 128 + 255) / 256, 256, 0, stream>>>(W_a,  Wt_a,  128, 128);
    kan_wprep<<<(128 * 128 + 255) / 256, 256, 0, stream>>>(W_l,  Wt_l,  128, 128);

    const int tilesN = (N + 15) / 16;
    const int tilesE = (E + 15) / 16;
    const int blkN = (tilesN + 3) / 4;
    const int blkE = (tilesE + 3) / 4;

    kan_node_pre<<<blkN, 128, 0, stream>>>(agts, Wt_a, a0, N);

    kan_edge<<<blkE, 128, 0, stream>>>(
        agts, ctx, agt_ctrs, ctx_ctrs, hi, wi,
        W_d1, b_d1, Wt_d2, g_d2, b_d2,
        Wt_q, g_q, b_q, Wt_c1, g_c1, b_c1, Wt_c2,
        a0, E);

    kan_node_post<<<blkN, 128, 0, stream>>>(
        agts, a0, Wt_l, g_n, b_n, g_l, b_l, (float*)d_out, N);
}
